// MultiLayerMHDSRA2Model_67851893342757
// MI455X (gfx1250) — matrix operand order repaired and checked by reading, not compiled, not run on hardware
//
#include <hip/hip_runtime.h>
#include <hip/hip_bf16.h>

// ---------------- model constants ----------------
#define VOCAB   32000
#define DIM     1024
#define HEADS   8
#define DH      128
#define LAYERS  2
#define K_SLOTS 128
#define TOPK    16      // KR == KW == 16
#define CHUNK   256
#define NCHUNK  16      // S / CHUNK
#define BATCH   2
#define SEQ     4096
#define GAMMA   0.9f
#define SCALE   0.08838834764831845f   // 1/sqrt(128)

typedef __attribute__((ext_vector_type(16))) __bf16 bf16x16;
typedef __attribute__((ext_vector_type(8)))  __bf16 bf16x8;
typedef __attribute__((ext_vector_type(4)))  __bf16 bf16x4;
typedef __attribute__((ext_vector_type(8)))  float  floatx8;

// GEMM tiling
#define BM 128          // block tile M
#define BN 128          // block tile N
#define BK 32           // K step (one WMMA depth)
#define LDA 40          // padded LDS row length (bf16): 80B stride, 16B aligned

static __device__ inline bf16x4 cvt4(float4 f)
{
    bf16x4 r;
    r[0] = (__bf16)f.x; r[1] = (__bf16)f.y;
    r[2] = (__bf16)f.z; r[3] = (__bf16)f.w;
    return r;
}

// =====================================================================
// LDS-tiled batched strided GEMM, bf16 WMMA, f32 accumulate.
//   C[z,m,n] = alpha * sum_k A[z,m,k]*B[z,k,n]  (+ Cin[z,m,n]) (+ bias[n])
// A is K-contiguous (element (m,k) at A[m*sAm + k]) at every call site.
// BPATH = 0 : B is N-contiguous,  element (k,n) at B[k*sBs + n]
// BPATH = 1 : B is K-contiguous,  element (k,n) at B[n*sBs + k]  (q.k^T)
// Block: 256 threads = 8 waves; block tile 128x128; each wave computes a
// 32x64 sub-tile (2x4 accumulators) -> 8 v_wmma per K-step per wave.
// Fills use float4 global loads + bf16x4 LDS stores, software-pipelined
// one K-panel ahead. Requires M%128==0, N%128==0, K%32==0 (true at every
// call site: M in {256,8192}, N in {1024,256,128,32000}, K in {1024,256,128}).
// =====================================================================
template<int BPATH>
__global__ void __launch_bounds__(256)
wmma_gemm_bf16(const float* __restrict__ A, const float* __restrict__ Bm,
               const float* __restrict__ Cin, const float* __restrict__ bias,
               float* __restrict__ C, float alpha,
               int M, int N, int K, int Hdiv,
               long sAo, long sAi, long sAm,
               long sBo, long sBi, long sBs,
               long sCo, long sCi, long sCm, long sCn)
{
    const int z  = blockIdx.z;
    const int zo = z / Hdiv, zi = z % Hdiv;
    const float* Ab = A  + (long)zo * sAo + (long)zi * sAi;
    const float* Bb = Bm + (long)zo * sBo + (long)zi * sBi;
    const long   coff = (long)zo * sCo + (long)zi * sCi;

    const int tid  = threadIdx.x;
    const int wave = tid >> 5;
    const int lane = tid & 31;
    const int lm   = lane & 15;     // M (A frag) / N (B,C frag) within 16-tile
    const int hi   = lane >> 4;     // lane half select

    const int wm = wave >> 1;       // 0..3 : wave row (32 rows each)
    const int wn = wave & 1;        // 0..1 : wave col (64 cols each)

    const int tm0 = blockIdx.y * BM;
    const int tn0 = blockIdx.x * BN;

    __shared__ __align__(16) __bf16 As[BM][LDA];   // [m][k]
    __shared__ __align__(16) __bf16 Bs[BN][LDA];   // [n][k]  (pre-transposed)

    // ---- per-thread fill geometry (constant across K loop) ----
    // A: 128x32 floats = 1024 float4; 4 per thread; coalesced along k
    int aRow[4], aK4[4];
    const float* aBase[4];
    #pragma unroll
    for (int it = 0; it < 4; ++it) {
        const int vid = tid + it * 256;
        aRow[it] = vid >> 3;                  // 0..127
        aK4[it]  = (vid & 7) * 4;             // 0..28
        aBase[it] = Ab + (long)(tm0 + aRow[it]) * sAm + aK4[it];
    }
    // B: 32x128 floats = 1024 float4; 4 per thread
    int bI0[4], bI1[4];                       // path0: (k, n4); path1: (n, k4)
    const float* bBase[4];
    #pragma unroll
    for (int it = 0; it < 4; ++it) {
        const int vid = tid + it * 256;
        if (BPATH == 0) {
            bI0[it] = vid >> 5;               // k   : 0..31
            bI1[it] = (vid & 31) * 4;         // n4  : 0..124
            bBase[it] = Bb + (long)bI0[it] * sBs + (tn0 + bI1[it]);
        } else {
            bI0[it] = vid >> 3;               // n   : 0..127
            bI1[it] = (vid & 7) * 4;          // k4  : 0..28
            bBase[it] = Bb + (long)(tn0 + bI0[it]) * sBs + bI1[it];
        }
    }

    floatx8 acc[2][4] = {};
    float4 aR[4], bR[4];

    // ---- prologue: stage K-panel 0 ----
    #pragma unroll
    for (int it = 0; it < 4; ++it) aR[it] = *(const float4*)(aBase[it]);
    #pragma unroll
    for (int it = 0; it < 4; ++it) bR[it] = *(const float4*)(bBase[it]);

    for (int kk = 0; kk < K; kk += BK) {
        __syncthreads();                       // LDS free (prev compute done)
        // ---- commit staged registers to LDS (bf16) ----
        #pragma unroll
        for (int it = 0; it < 4; ++it)
            *(bf16x4*)&As[aRow[it]][aK4[it]] = cvt4(aR[it]);
        #pragma unroll
        for (int it = 0; it < 4; ++it) {
            if (BPATH == 0) {
                const bf16x4 bv = cvt4(bR[it]);
                const int k = bI0[it], n4 = bI1[it];
                Bs[n4 + 0][k] = bv[0];
                Bs[n4 + 1][k] = bv[1];
                Bs[n4 + 2][k] = bv[2];
                Bs[n4 + 3][k] = bv[3];
            } else {
                *(bf16x4*)&Bs[bI0[it]][bI1[it]] = cvt4(bR[it]);
            }
        }
        __syncthreads();                       // tile visible to all waves

        // ---- software pipeline: stage next K-panel while computing ----
        if (kk + BK < K) {
            const long ka = kk + BK;
            #pragma unroll
            for (int it = 0; it < 4; ++it)
                aR[it] = *(const float4*)(aBase[it] + ka);
            #pragma unroll
            for (int it = 0; it < 4; ++it)
                bR[it] = (BPATH == 0) ? *(const float4*)(bBase[it] + ka * sBs)
                                      : *(const float4*)(bBase[it] + ka);
        }

        // ---- fragments from LDS + 8 WMMA per wave ----
        bf16x16 afrag[2], bfrag[4];
        #pragma unroll
        for (int am = 0; am < 2; ++am) {
            const int row = wm * 32 + am * 16 + lm;
            // A layout: lane half 0 -> K {0..7, 16..23}; half 1 -> {8..15, 24..31}
            bf16x8 a0 = *(const bf16x8*)&As[row][hi * 8];
            bf16x8 a1 = *(const bf16x8*)&As[row][16 + hi * 8];
            afrag[am] = __builtin_shufflevector(a0, a1,
                          0,1,2,3,4,5,6,7,8,9,10,11,12,13,14,15);
        }
        #pragma unroll
        for (int bn = 0; bn < 4; ++bn) {
            const int rowN = wn * 64 + bn * 16 + lm;
            // B layout: lane half 0 -> K {0..7, 16..23}; half 1 -> {8..15, 24..31}
            bf16x8 b0 = *(const bf16x8*)&Bs[rowN][hi * 8];
            bf16x8 b1 = *(const bf16x8*)&Bs[rowN][16 + hi * 8];
            bfrag[bn] = __builtin_shufflevector(b0, b1,
                          0,1,2,3,4,5,6,7,8,9,10,11,12,13,14,15);
        }
        #pragma unroll
        for (int am = 0; am < 2; ++am)
            #pragma unroll
            for (int bn = 0; bn < 4; ++bn)
                acc[am][bn] = __builtin_amdgcn_wmma_f32_16x16x32_bf16(
                    false, afrag[am], false, bfrag[bn], (short)0,
                    acc[am][bn], false, false);
    }

    // ---- epilogue: C/D layout: lane lm = N; VGPR r -> M = r + 8*hi ----
    #pragma unroll
    for (int am = 0; am < 2; ++am) {
        #pragma unroll
        for (int bn = 0; bn < 4; ++bn) {
            #pragma unroll
            for (int r = 0; r < 8; ++r) {
                const int m = tm0 + wm * 32 + am * 16 + r + hi * 8;
                const int n = tn0 + wn * 64 + bn * 16 + lm;
                if (m < M && n < N) {
                    const long off = coff + (long)m * sCm + (long)n * sCn;
                    float val = alpha * acc[am][bn][r];
                    if (Cin)  val += Cin[off];
                    if (bias) val += bias[n];
                    C[off] = val;
                }
            }
        }
    }
}

// =====================================================================
// Embedding gather: hidden[i,:] = embed[tokens[i],:]
// =====================================================================
__global__ void embed_gather(const int* __restrict__ tok,
                             const float* __restrict__ emb,
                             float* __restrict__ out)
{
    const long i = blockIdx.x;
    const float* e = emb + (long)tok[i] * DIM;
    float* o = out + i * DIM;
    for (int j = threadIdx.x; j < DIM; j += blockDim.x) o[j] = e[j];
}

// =====================================================================
// LayerNorm over rows of length DIM. Row r -> (b = r/rowsPerB, t = r%rowsPerB)
// input at x + b*inB + t*DIM, output at y + b*outB + t*DIM. Safe in-place.
// =====================================================================
__global__ void __launch_bounds__(256)
layernorm_rows(const float* __restrict__ x, float* __restrict__ y,
               const float* __restrict__ g, const float* __restrict__ bsh,
               int rowsPerB, long inB, long outB)
{
    const int r  = blockIdx.x;
    const int bb = r / rowsPerB, t = r % rowsPerB;
    const float* xi = x + (long)bb * inB + (long)t * DIM;
    float*       yo = y + (long)bb * outB + (long)t * DIM;
    __shared__ float red[256];

    float s = 0.f;
    for (int j = threadIdx.x; j < DIM; j += 256) s += xi[j];
    red[threadIdx.x] = s; __syncthreads();
    for (int o = 128; o > 0; o >>= 1) {
        if (threadIdx.x < o) red[threadIdx.x] += red[threadIdx.x + o];
        __syncthreads();
    }
    const float mu = red[0] * (1.0f / DIM);
    __syncthreads();

    float s2 = 0.f;
    for (int j = threadIdx.x; j < DIM; j += 256) { float d = xi[j] - mu; s2 += d * d; }
    red[threadIdx.x] = s2; __syncthreads();
    for (int o = 128; o > 0; o >>= 1) {
        if (threadIdx.x < o) red[threadIdx.x] += red[threadIdx.x + o];
        __syncthreads();
    }
    const float inv = rsqrtf(red[0] * (1.0f / DIM) + 1e-5f);

    for (int j = threadIdx.x; j < DIM; j += 256)
        yo[j] = (xi[j] - mu) * inv * g[j] + bsh[j];
}

// =====================================================================
// Causal softmax over att rows. blockIdx.x = (b*H + h)*CHUNK + t, block=CHUNK.
// =====================================================================
__global__ void __launch_bounds__(CHUNK)
causal_softmax(float* __restrict__ att)
{
    const int t = blockIdx.x % CHUNK;
    float* row = att + (long)blockIdx.x * CHUNK;
    const int s = threadIdx.x;
    const float x = (s <= t) ? row[s] : -1e30f;
    __shared__ float red[CHUNK];
    red[s] = x; __syncthreads();
    for (int o = CHUNK / 2; o > 0; o >>= 1) {
        if (s < o) red[s] = fmaxf(red[s], red[s + o]);
        __syncthreads();
    }
    const float mx = red[0]; __syncthreads();
    const float e = (s <= t) ? __expf(x - mx) : 0.f;
    red[s] = e; __syncthreads();
    for (int o = CHUNK / 2; o > 0; o >>= 1) {
        if (s < o) red[s] += red[s + o];
        __syncthreads();
    }
    row[s] = e / red[0];
}

// =====================================================================
// Slot READ: per (b,h,t) block of 128 threads.
// =====================================================================
__global__ void __launch_bounds__(K_SLOTS)
slot_read(const float* __restrict__ q, const float* __restrict__ sk,
          const float* __restrict__ Mslots, float* __restrict__ readD)
{
    const int idx = blockIdx.x;
    const int t = idx % CHUNK;
    const int h = (idx / CHUNK) % HEADS;
    const int b = idx / (CHUNK * HEADS);
    const int tid = threadIdx.x;

    __shared__ float qs[DH];
    __shared__ float val[K_SLOTS];
    __shared__ float rv[K_SLOTS];
    __shared__ int   ri[K_SLOTS];
    __shared__ float tw[TOPK];
    __shared__ int   ti[TOPK];

    qs[tid] = q[((long)b * CHUNK + t) * DIM + h * DH + tid];
    __syncthreads();

    const float* skr = sk + ((long)h * K_SLOTS + tid) * DH;
    float sc = 0.f;
    #pragma unroll 4
    for (int d = 0; d < DH; ++d) sc += qs[d] * skr[d];
    val[tid] = sc * SCALE;
    __syncthreads();

    for (int j = 0; j < TOPK; ++j) {
        rv[tid] = val[tid]; ri[tid] = tid; __syncthreads();
        for (int o = K_SLOTS / 2; o > 0; o >>= 1) {
            if (tid < o && rv[tid + o] > rv[tid]) { rv[tid] = rv[tid + o]; ri[tid] = ri[tid + o]; }
            __syncthreads();
        }
        if (tid == 0) { tw[j] = rv[0]; ti[j] = ri[0]; val[ri[0]] = -3e38f; }
        __syncthreads();
    }
    if (tid < TOPK) tw[tid] = __expf(tw[tid] - tw[0]);
    __syncthreads();
    if (tid == 0) { float s = 0.f; for (int j = 0; j < TOPK; ++j) s += tw[j]; rv[0] = 1.f / s; }
    __syncthreads();
    const float inv = rv[0];

    float acc = 0.f;
    const float* Mb = Mslots + ((long)b * HEADS + h) * K_SLOTS * DH;
    for (int j = 0; j < TOPK; ++j)
        acc += tw[j] * inv * Mb[(long)ti[j] * DH + tid];
    readD[((long)b * CHUNK + t) * DIM + h * DH + tid] = acc;
}

// =====================================================================
// Slot memory decay: M *= GAMMA
// =====================================================================
__global__ void decay_mem(float* __restrict__ M, int n)
{
    const int i = blockIdx.x * blockDim.x + threadIdx.x;
    if (i < n) M[i] *= GAMMA;
}

// =====================================================================
// Slot WRITE: top-16(k.sk) softmax -> atomic scatter into M (post-decay)
// =====================================================================
__global__ void __launch_bounds__(K_SLOTS)
slot_write(const float* __restrict__ kten, const float* __restrict__ sk,
           const float* __restrict__ vten, float* __restrict__ Mslots)
{
    const int idx = blockIdx.x;
    const int t = idx % CHUNK;
    const int h = (idx / CHUNK) % HEADS;
    const int b = idx / (CHUNK * HEADS);
    const int tid = threadIdx.x;

    __shared__ float ks[DH];
    __shared__ float val[K_SLOTS];
    __shared__ float rv[K_SLOTS];
    __shared__ int   ri[K_SLOTS];
    __shared__ float tw[TOPK];
    __shared__ int   ti[TOPK];

    ks[tid] = kten[((long)b * CHUNK + t) * DIM + h * DH + tid];
    __syncthreads();

    const float* skr = sk + ((long)h * K_SLOTS + tid) * DH;
    float sc = 0.f;
    #pragma unroll 4
    for (int d = 0; d < DH; ++d) sc += ks[d] * skr[d];
    val[tid] = sc * SCALE;
    __syncthreads();

    for (int j = 0; j < TOPK; ++j) {
        rv[tid] = val[tid]; ri[tid] = tid; __syncthreads();
        for (int o = K_SLOTS / 2; o > 0; o >>= 1) {
            if (tid < o && rv[tid + o] > rv[tid]) { rv[tid] = rv[tid + o]; ri[tid] = ri[tid + o]; }
            __syncthreads();
        }
        if (tid == 0) { tw[j] = rv[0]; ti[j] = ri[0]; val[ri[0]] = -3e38f; }
        __syncthreads();
    }
    if (tid < TOPK) tw[tid] = __expf(tw[tid] - tw[0]);
    __syncthreads();
    if (tid == 0) { float s = 0.f; for (int j = 0; j < TOPK; ++j) s += tw[j]; rv[0] = 1.f / s; }
    __syncthreads();
    const float inv = rv[0];

    const float vd = vten[((long)b * CHUNK + t) * DIM + h * DH + tid];
    float* Mb = Mslots + ((long)b * HEADS + h) * K_SLOTS * DH;
    for (int j = 0; j < TOPK; ++j)
        atomicAdd(&Mb[(long)ti[j] * DH + tid], tw[j] * inv * vd);
}

// =====================================================================
// Host-side GEMM launch helper. bpath: 0 = B N-contiguous, 1 = B K-contiguous.
// =====================================================================
static inline void launch_gemm(hipStream_t st, int bpath,
                               const float* A, const float* B,
                               const float* Cin, const float* bias, float* C,
                               float alpha, int M, int N, int K, int batch, int Hdiv,
                               long sAo, long sAi, long sAm,
                               long sBo, long sBi, long sBs,
                               long sCo, long sCi, long sCm, long sCn)
{
    dim3 grid((N + BN - 1) / BN, (M + BM - 1) / BM, batch);
    if (bpath == 0)
        wmma_gemm_bf16<0><<<grid, 256, 0, st>>>(A, B, Cin, bias, C, alpha, M, N, K, Hdiv,
                                                sAo, sAi, sAm, sBo, sBi, sBs,
                                                sCo, sCi, sCm, sCn);
    else
        wmma_gemm_bf16<1><<<grid, 256, 0, st>>>(A, B, Cin, bias, C, alpha, M, N, K, Hdiv,
                                                sAo, sAi, sAm, sBo, sBi, sBs,
                                                sCo, sCi, sCm, sCn);
}

extern "C" void kernel_launch(void* const* d_in, const int* in_sizes, int n_in,
                              void* d_out, int out_size, void* d_ws, size_t ws_size,
                              hipStream_t stream)
{
    (void)in_sizes; (void)n_in; (void)out_size; (void)ws_size;

    const int*   tokens    = (const int*)  d_in[0];
    const float* embed     = (const float*)d_in[1];
    const float* Wq        = (const float*)d_in[2];
    const float* Wk        = (const float*)d_in[3];
    const float* Wv        = (const float*)d_in[4];
    const float* Wo        = (const float*)d_in[5];
    const float* slot_keys = (const float*)d_in[6];
    const float* ln_g      = (const float*)d_in[7];
    const float* ln_b      = (const float*)d_in[8];
    const float* fg        = (const float*)d_in[9];
    const float* fb        = (const float*)d_in[10];
    const float* Wout      = (const float*)d_in[11];
    const float* bout      = (const float*)d_in[12];
    float*       out       = (float*)d_out;

    // ---- workspace carve-up (floats) ----
    float* ws = (float*)d_ws;
    float* hidden = ws;                                   // [B,S,D]
    float* xn     = hidden + (long)BATCH * SEQ * DIM;     // [B,T,D]
    float* qb     = xn     + (long)BATCH * CHUNK * DIM;   // [B,T,D]
    float* kb     = qb     + (long)BATCH * CHUNK * DIM;   // [B,T,D]
    float* vb     = kb     + (long)BATCH * CHUNK * DIM;   // [B,T,D]
    float* readD  = vb     + (long)BATCH * CHUNK * DIM;   // [B,T,D]
    float* bufD   = readD  + (long)BATCH * CHUNK * DIM;   // [B,T,D]
    float* att    = bufD   + (long)BATCH * CHUNK * DIM;   // [B,H,T,T]
    float* Mbuf   = att    + (long)BATCH * HEADS * CHUNK * CHUNK; // [L,B,H,K,dh]

    const long Msz = (long)LAYERS * BATCH * HEADS * K_SLOTS * DH;
    hipMemsetAsync(Mbuf, 0, Msz * sizeof(float), stream);

    // ---- embedding gather ----
    embed_gather<<<BATCH * SEQ, 256, 0, stream>>>(tokens, embed, hidden);

    const long sBD = (long)SEQ * DIM;     // b-stride in hidden
    const long tBD = (long)CHUNK * DIM;   // b-stride in chunk buffers

    for (int c = 0; c < NCHUNK; ++c) {
        float* hid_chunk = hidden + (long)c * CHUNK * DIM;   // [:, c*T:(c+1)*T, :]
        for (int l = 0; l < LAYERS; ++l) {
            const float* Wq_l = Wq + (long)l * DIM * DIM;
            const float* Wk_l = Wk + (long)l * DIM * DIM;
            const float* Wv_l = Wv + (long)l * DIM * DIM;
            const float* Wo_l = Wo + (long)l * DIM * DIM;
            const float* sk_l = slot_keys + (long)l * HEADS * K_SLOTS * DH;
            float*       M_l  = Mbuf + (long)l * BATCH * HEADS * K_SLOTS * DH;

            // LayerNorm(chunk) -> xn  [B,T,D]
            layernorm_rows<<<BATCH * CHUNK, 256, 0, stream>>>(
                hid_chunk, xn, ln_g + (long)l * DIM, ln_b + (long)l * DIM,
                CHUNK, sBD, tBD);

            // q,k,v = xn @ W  (batch over B, M=T, K=N=DIM; B N-contig)
            launch_gemm(stream, 0, xn, Wq_l, nullptr, nullptr, qb, 1.f,
                        CHUNK, DIM, DIM, BATCH, 1,
                        tBD, 0, DIM,   0, 0, DIM,   tBD, 0, DIM, 1);
            launch_gemm(stream, 0, xn, Wk_l, nullptr, nullptr, kb, 1.f,
                        CHUNK, DIM, DIM, BATCH, 1,
                        tBD, 0, DIM,   0, 0, DIM,   tBD, 0, DIM, 1);
            launch_gemm(stream, 0, xn, Wv_l, nullptr, nullptr, vb, 1.f,
                        CHUNK, DIM, DIM, BATCH, 1,
                        tBD, 0, DIM,   0, 0, DIM,   tBD, 0, DIM, 1);

            // slot READ (uses M before update)
            slot_read<<<BATCH * HEADS * CHUNK, K_SLOTS, 0, stream>>>(
                qb, sk_l, M_l, readD);

            // attention scores: att[b,h] = scale * q[b,h] @ k[b,h]^T
            // B = k^T is K-contiguous: element (k=d, n=s) at kb[s*DIM + d]
            launch_gemm(stream, 1, qb, kb, nullptr, nullptr, att, SCALE,
                        CHUNK, CHUNK, DH, BATCH * HEADS, HEADS,
                        tBD, DH, DIM,
                        tBD, DH, DIM,
                        (long)HEADS * CHUNK * CHUNK, (long)CHUNK * CHUNK, CHUNK, 1);

            causal_softmax<<<BATCH * HEADS * CHUNK, CHUNK, 0, stream>>>(att);

            // bufD = att @ v + read   (B = v is N-contig; write in [B,T,D])
            launch_gemm(stream, 0, att, vb, readD, nullptr, bufD, 1.f,
                        CHUNK, DH, CHUNK, BATCH * HEADS, HEADS,
                        (long)HEADS * CHUNK * CHUNK, (long)CHUNK * CHUNK, CHUNK,
                        tBD, DH, DIM,
                        tBD, DH, DIM, 1);

            // chunk = chunk + bufD @ Wo   (residual fused via Cin)
            launch_gemm(stream, 0, bufD, Wo_l, hid_chunk, nullptr, hid_chunk, 1.f,
                        CHUNK, DIM, DIM, BATCH, 1,
                        tBD, 0, DIM,   0, 0, DIM,   sBD, 0, DIM, 1);

            // slot WRITE: M = GAMMA*M + scatter(top-16(k.sk), v)
            const int nM = BATCH * HEADS * K_SLOTS * DH;
            decay_mem<<<(nM + 255) / 256, 256, 0, stream>>>(M_l, nM);
            slot_write<<<BATCH * HEADS * CHUNK, K_SLOTS, 0, stream>>>(
                kb, sk_l, vb, M_l);
        }
    }

    // final LayerNorm (in-place on hidden)
    layernorm_rows<<<BATCH * SEQ, 256, 0, stream>>>(
        hidden, hidden, fg, fb, SEQ, sBD, sBD);

    // logits = hidden @ Wout + bout   [8192 x 32000], the FLOP-dominant GEMM
    launch_gemm(stream, 0, hidden, Wout, nullptr, bout, out, 1.f,
                BATCH * SEQ, VOCAB, DIM, 1, 1,
                0, 0, DIM,   0, 0, VOCAB,   0, 0, VOCAB, 1);
}
